// MoeLayer_34737695490106
// MI455X (gfx1250) — matrix operand order repaired and checked by reading, not compiled, not run on hardware
//
#include <hip/hip_runtime.h>
#include <stdint.h>

#define EMB 1024
#define HID 4096
#define NE 8
#define TTOK 8192
#define CAP 8192

typedef unsigned short u16t;
typedef __attribute__((ext_vector_type(16))) __bf16 v16bf;
typedef __attribute__((ext_vector_type(8)))  float  v8f;

union Frag { uint4 u4[2]; v16bf v; };

__device__ __forceinline__ u16t f2bf(float f) {
    unsigned u = __float_as_uint(f);
    u += 0x7fffu + ((u >> 16) & 1u);          // round-to-nearest-even
    return (u16t)(u >> 16);
}
__device__ __forceinline__ float bf2f(u16t h) {
    return __uint_as_float(((unsigned)h) << 16);
}

// CDNA5 async global->LDS copy (ASYNCcnt tracked, no VGPR staging).
__device__ __forceinline__ void async_b128(uint32_t lds_addr, const void* gaddr) {
    asm volatile("global_load_async_to_lds_b128 %0, %1, off"
                 :: "v"(lds_addr), "v"(gaddr) : "memory");
}
__device__ __forceinline__ void wait_async0() {
    asm volatile("s_wait_asynccnt 0x0" ::: "memory");
}

// ---------------------------------------------------------------- init ------
__global__ __launch_bounds__(256) void k_init(float4* __restrict__ out4, int n4,
                                              int* __restrict__ cnt) {
    int i = blockIdx.x * 256 + threadIdx.x;
    if (i < n4) out4[i] = make_float4(0.f, 0.f, 0.f, 0.f);
    if (i < NE) cnt[i] = 0;
}

// ------------------------------------------------------------- f32->bf16 ----
__global__ __launch_bounds__(256) void k_convert(const float4* __restrict__ in,
                                                 uint2* __restrict__ out, int n4) {
    int i = blockIdx.x * 256 + threadIdx.x;
    if (i >= n4) return;
    float4 v = in[i];
    out[i] = make_uint2((unsigned)f2bf(v.x) | ((unsigned)f2bf(v.y) << 16),
                        (unsigned)f2bf(v.z) | ((unsigned)f2bf(v.w) << 16));
}

// ------------------------------------- f32 [B][R][C] -> bf16 [B][C][R] ------
__global__ __launch_bounds__(256) void k_transpose(const float* __restrict__ in,
                                                   u16t* __restrict__ out,
                                                   int R, int C) {
    __shared__ float tile[32][33];
    size_t b = (size_t)blockIdx.z * (size_t)R * (size_t)C;
    int r0 = blockIdx.y * 32, c0 = blockIdx.x * 32;
    int tx = threadIdx.x & 31, ty = threadIdx.x >> 5;
    for (int i = 0; i < 32; i += 8)
        tile[ty + i][tx] = in[b + (size_t)(r0 + ty + i) * C + (c0 + tx)];
    __syncthreads();
    for (int i = 0; i < 32; i += 8)
        out[b + (size_t)(c0 + ty + i) * R + (r0 + tx)] = f2bf(tile[tx][ty + i]);
}

// ------------------------------------------------------ WMMA tiled GEMM -----
// MODE 0: Hr = silu(xb @ rw1t^T + rb1)                 (router MLP1)
// MODE 1: Hx[off[e]+i] = silu(xb[tok] @ w1t[e]^T + b1) (gathered expert MLP1)
// MODE 2: out[tok] += we * (Hx @ w2t[e]^T + b2)        (expert MLP2, scatter)
//
// 128x128 C tile, BK=64, 8 waves; double-buffered LDS fed by async-to-LDS.
template <int MODE>
__global__ __launch_bounds__(256) void moe_gemm(
    const u16t* __restrict__ A, const u16t* __restrict__ Bt,
    const float* __restrict__ bias, u16t* __restrict__ Hout,
    float* __restrict__ Fout,
    const int* __restrict__ cnt, const int* __restrict__ off,
    const int* __restrict__ tok_id, const float* __restrict__ tok_w) {

    constexpr int KK = (MODE == 2) ? HID : EMB;
    constexpr int NN = (MODE == 2) ? EMB : HID;
    constexpr int BK = 64;
    constexpr int LS = BK + 8;          // padded LDS row stride (ushorts)
    constexpr int NSTEP = KK / BK;

    const int e  = blockIdx.z;
    const int m0 = blockIdx.y * 128;
    const int n0 = blockIdx.x * 128;

    int rows = TTOK, offE = 0;
    const int*   tidE = nullptr;
    const float* twE  = nullptr;
    if constexpr (MODE != 0) {
        rows = cnt[e];
        if (m0 >= rows) return;         // uniform early exit (empty tile)
        offE = off[e];
        tidE = tok_id + e * CAP;
        twE  = tok_w + e * CAP;
    }
    const u16t*  BtE   = Bt + (size_t)e * NN * KK;
    const float* biasE = bias + (size_t)e * NN;

    __shared__ u16t lA[2][128 * LS];
    __shared__ u16t lB[2][128 * LS];

    // per-thread global->LDS descriptors: 4 16B chunks for A, 4 for B per tile
    // chunk c = tid + i*256 -> row = c>>3, kc = (c&7)*8
    int aRow[4]; int kcv[4]; int bCol[4];
#pragma unroll
    for (int i = 0; i < 4; i++) {
        int c = threadIdx.x + i * 256;
        int row = c >> 3;
        kcv[i]  = (c & 7) * 8;
        bCol[i] = row;
        if constexpr (MODE == 0) {
            aRow[i] = m0 + row;
        } else if constexpr (MODE == 1) {
            int rr = m0 + row; if (rr >= rows) rr = rows - 1;   // clamp: keep EXEC full
            aRow[i] = tidE[rr];
        } else {
            int rr = m0 + row; if (rr >= rows) rr = rows - 1;
            aRow[i] = offE + rr;
        }
    }

    auto issue = [&](int k0, int buf) {
#pragma unroll
        for (int i = 0; i < 4; i++) {
            uint32_t la = (uint32_t)(uintptr_t)&lA[buf][bCol[i] * LS + kcv[i]];
            async_b128(la, A + (size_t)aRow[i] * KK + k0 + kcv[i]);
            uint32_t lb = (uint32_t)(uintptr_t)&lB[buf][bCol[i] * LS + kcv[i]];
            async_b128(lb, BtE + (size_t)(n0 + bCol[i]) * KK + k0 + kcv[i]);
        }
    };

    const int w    = threadIdx.x >> 5;
    const int lane = threadIdx.x & 31;
    const int half = lane >> 4;
    const int lm   = lane & 15;

    v8f zero = {0.f, 0.f, 0.f, 0.f, 0.f, 0.f, 0.f, 0.f};
    v8f acc[8];
#pragma unroll
    for (int cb = 0; cb < 8; cb++) acc[cb] = zero;

    issue(0, 0);
    int cur = 0;
    for (int s = 0; s < NSTEP; s++) {
        wait_async0();          // this wave's async writes to buf[cur] landed
        __syncthreads();        // all waves landed; prev reads of buf[cur^1] done
        if (s + 1 < NSTEP) issue((s + 1) * BK, cur ^ 1);   // prefetch behind compute

#pragma unroll
        for (int ks = 0; ks < 2; ks++) {
            // A fragment 16x32: lanes 0-15 K 0..15, lanes 16-31 K 16..31
            Frag fa;
            const u16t* ap = &lA[cur][(w * 16 + lm) * LS + ks * 32 + half * 16];
            fa.u4[0] = *(const uint4*)(ap);
            fa.u4[1] = *(const uint4*)(ap + 8);
            // B fragments 32x16: lanes 0-15 K 0..15, lanes 16-31 K 16..31
            Frag fb[8];
#pragma unroll
            for (int cb = 0; cb < 8; cb++) {
                const u16t* bp = &lB[cur][(cb * 16 + lm) * LS + ks * 32 + half * 16];
                fb[cb].u4[0] = *(const uint4*)(bp);
                fb[cb].u4[1] = *(const uint4*)(bp + 8);
            }
#pragma unroll
            for (int cb = 0; cb < 8; cb++)
                acc[cb] = __builtin_amdgcn_wmma_f32_16x16x32_bf16(
                    false, fa.v, false, fb[cb].v, (short)0, acc[cb], false, false);
        }
        cur ^= 1;
    }

    // epilogue: C/D layout -> row = w*16 + half*8 + r, col = n0 + cb*16 + lm
#pragma unroll
    for (int cb = 0; cb < 8; cb++) {
        int col = n0 + cb * 16 + lm;
#pragma unroll
        for (int r = 0; r < 8; r++) {
            int rloc = w * 16 + half * 8 + r;
            float v = acc[cb][r] + biasE[col];
            if constexpr (MODE == 0) {
                v = v / (1.f + __expf(-v));
                Hout[(size_t)(m0 + rloc) * NN + col] = f2bf(v);
            } else if constexpr (MODE == 1) {
                if (m0 + rloc < rows) {
                    v = v / (1.f + __expf(-v));
                    Hout[(size_t)(offE + m0 + rloc) * NN + col] = f2bf(v);
                }
            } else {
                if (m0 + rloc < rows) {
                    int tok = tidE[m0 + rloc];
                    float we = twE[m0 + rloc];
                    atomicAdd(&Fout[(size_t)tok * NN + col], v * we);
                }
            }
        }
    }
}

// --------------------------------------------- router logits + top2 ---------
__global__ __launch_bounds__(256) void k_router(const u16t* __restrict__ Hr,
                                                const float* __restrict__ rw2,
                                                const float* __restrict__ rb2,
                                                int* __restrict__ cnt,
                                                int* __restrict__ tok_id,
                                                float* __restrict__ tok_w) {
    int wv   = (blockIdx.x * 256 + threadIdx.x) >> 5;  // one wave32 per token
    int lane = threadIdx.x & 31;
    const u16t* h = Hr + (size_t)wv * HID;

    float lg[NE];
#pragma unroll
    for (int e = 0; e < NE; e++) lg[e] = 0.f;

    for (int base = lane * 8; base < HID; base += 256) {
        uint4 hv = *(const uint4*)(h + base);
        const u16t* hp = (const u16t*)&hv;
#pragma unroll
        for (int j = 0; j < 8; j++) {
            float hf = bf2f(hp[j]);
            const float4* wr = (const float4*)(rw2 + (size_t)(base + j) * NE);
            float4 w0 = wr[0], w1 = wr[1];
            lg[0] += hf * w0.x; lg[1] += hf * w0.y;
            lg[2] += hf * w0.z; lg[3] += hf * w0.w;
            lg[4] += hf * w1.x; lg[5] += hf * w1.y;
            lg[6] += hf * w1.z; lg[7] += hf * w1.w;
        }
    }
#pragma unroll
    for (int e = 0; e < NE; e++)
        for (int s = 16; s > 0; s >>= 1) lg[e] += __shfl_xor(lg[e], s, 32);

    if (lane == 0) {
        float lv[NE];
#pragma unroll
        for (int e = 0; e < NE; e++) lv[e] = lg[e] + rb2[e];
        int i0 = 0; float l0 = lv[0];
#pragma unroll
        for (int e = 1; e < NE; e++) if (lv[e] > l0) { l0 = lv[e]; i0 = e; }
        int i1 = -1; float l1 = -3.4e38f;
#pragma unroll
        for (int e = 0; e < NE; e++)
            if (e != i0 && lv[e] > l1) { l1 = lv[e]; i1 = e; }
        float p1  = __expf(l1 - l0);
        float inv = 1.f / (1.f + p1);
        float p0  = inv;  p1 *= inv;
        int s0 = atomicAdd(&cnt[i0], 1);
        tok_id[i0 * CAP + s0] = wv; tok_w[i0 * CAP + s0] = p0;
        int s1 = atomicAdd(&cnt[i1], 1);
        tok_id[i1 * CAP + s1] = wv; tok_w[i1 * CAP + s1] = p1;
    }
}

// ------------------------------------------------- prefix offsets (8) -------
__global__ void k_offsets(const int* __restrict__ cnt, int* __restrict__ off) {
    if (threadIdx.x == 0) {
        int a = 0;
        for (int e = 0; e < NE; e++) { off[e] = a; a += cnt[e]; }
    }
}

// ---------------------------------------------------------------- host ------
extern "C" void kernel_launch(void* const* d_in, const int* in_sizes, int n_in,
                              void* d_out, int out_size, void* d_ws, size_t ws_size,
                              hipStream_t stream) {
    (void)in_sizes; (void)n_in; (void)out_size; (void)ws_size;
    const float* x   = (const float*)d_in[0];
    const float* rw1 = (const float*)d_in[1];
    const float* rb1 = (const float*)d_in[2];
    const float* rw2 = (const float*)d_in[3];
    const float* rb2 = (const float*)d_in[4];
    const float* w1  = (const float*)d_in[5];
    const float* b1  = (const float*)d_in[6];
    const float* w2  = (const float*)d_in[7];
    const float* b2  = (const float*)d_in[8];
    float* out = (float*)d_out;

    char* ws = (char*)d_ws;
    size_t o = 0;
    auto alloc = [&](size_t bytes) -> void* {
        void* p = ws + o;
        o += (bytes + 255) & ~(size_t)255;
        return p;
    };
    u16t*  xb     = (u16t*)alloc((size_t)TTOK * EMB * 2);
    u16t*  rw1t   = (u16t*)alloc((size_t)HID * EMB * 2);
    u16t*  w1t    = (u16t*)alloc((size_t)NE * HID * EMB * 2);
    u16t*  w2t    = (u16t*)alloc((size_t)NE * EMB * HID * 2);
    u16t*  Hr     = (u16t*)alloc((size_t)TTOK * HID * 2);
    u16t*  Hx     = (u16t*)alloc((size_t)TTOK * 2 * HID * 2); // top-2 compacted
    int*   cnt    = (int*)alloc(NE * 4);
    int*   offs   = (int*)alloc(NE * 4);
    int*   tok_id = (int*)alloc((size_t)NE * CAP * 4);
    float* tok_w  = (float*)alloc((size_t)NE * CAP * 4);

    dim3 blk(256, 1, 1);
    const int n4 = TTOK * EMB / 4;

    k_init<<<dim3((n4 + 255) / 256, 1, 1), blk, 0, stream>>>((float4*)out, n4, cnt);
    k_convert<<<dim3(n4 / 256, 1, 1), blk, 0, stream>>>((const float4*)x, (uint2*)xb, n4);
    k_transpose<<<dim3(HID / 32, EMB / 32, 1),  blk, 0, stream>>>(rw1, rw1t, EMB, HID);
    k_transpose<<<dim3(HID / 32, EMB / 32, NE), blk, 0, stream>>>(w1,  w1t,  EMB, HID);
    k_transpose<<<dim3(EMB / 32, HID / 32, NE), blk, 0, stream>>>(w2,  w2t,  HID, EMB);

    moe_gemm<0><<<dim3(HID / 128, TTOK / 128, 1), blk, 0, stream>>>(
        xb, rw1t, rb1, Hr, nullptr, nullptr, nullptr, nullptr, nullptr);
    k_router<<<dim3(TTOK * 32 / 256, 1, 1), blk, 0, stream>>>(
        Hr, rw2, rb2, cnt, tok_id, tok_w);
    k_offsets<<<dim3(1, 1, 1), dim3(32, 1, 1), 0, stream>>>(cnt, offs);
    moe_gemm<1><<<dim3(HID / 128, TTOK / 128, NE), blk, 0, stream>>>(
        xb, w1t, b1, Hx, nullptr, cnt, offs, tok_id, tok_w);
    moe_gemm<2><<<dim3(EMB / 128, TTOK / 128, NE), blk, 0, stream>>>(
        Hx, w2t, b2, nullptr, out, cnt, offs, tok_id, tok_w);
}
